// RoutedDeltaGPT_88089779241167
// MI455X (gfx1250) — matrix operand order repaired and checked by reading, not compiled, not run on hardware
//
#include <hip/hip_runtime.h>
#include <hip/hip_bf16.h>
#include <math.h>

// ---------------------------------------------------------------------------
// RoutedDeltaGPT forward for MI455X (gfx1250, wave32, WMMA, async-LDS DMA).
//
// Strategy: one-time f32->bf16 conversion of all weights + GEMM activations
// (halves HBM weight traffic: ~370MB -> ~16us at 23.3 TB/s; weights are
// re-read by 16 M-blocks so L2 footprint halves too). GEMM hot loop then
// streams pure bf16 tiles into double-buffered LDS via
// global_load_async_to_lds_b128 (ASYNCcnt), no per-iteration conversion VALU,
// 8 v_wmma_f32_16x16x32_bf16 per wave per K-step with f32 accumulators.
// ---------------------------------------------------------------------------

typedef __attribute__((ext_vector_type(16))) __bf16 v16bf;
typedef __attribute__((ext_vector_type(8)))  float  v8f;
typedef __attribute__((ext_vector_type(4)))  int    v4i;

#define AS1 __attribute__((address_space(1)))
#define AS3 __attribute__((address_space(3)))

#if defined(__has_builtin)
#  if __has_builtin(__builtin_amdgcn_global_load_async_to_lds_b128)
#    define HAVE_ASYNC_LDS 1
#  else
#    define HAVE_ASYNC_LDS 0
#  endif
#else
#  define HAVE_ASYNC_LDS 0
#endif

constexpr int Bc   = 2;
constexpr int Tc   = 1024;
constexpr int Dc   = 768;
constexpr int Hc   = 12;
constexpr int Lc   = 4;
constexpr int Ec   = 8;
constexpr int DFFc = 3072;
constexpr int Vc   = 32000;
constexpr int HEADc = Dc / Hc;          // 64
constexpr int BT   = Bc * Tc;           // 2048

// GEMM tiling (bf16 operands)
constexpr int BM = 128;
constexpr int BN = 64;
constexpr int BK = 64;
constexpr int LDAs = 72;   // bf16 elems per LDS row: 64 + 8 pad (144B, 16B-mult, conflict-free)
constexpr int LDBs = 72;

enum { EPI_STORE = 0, EPI_RELU_BF16 = 1, EPI_ADD = 2, EPI_SCALE_ADD = 3 };

// ---- f32 <-> bf16 helpers -------------------------------------------------
__device__ inline __bf16 f2bf(float f) {
  union { float f; unsigned u; } v; v.f = f;
  unsigned r = v.u + 0x7FFFu + ((v.u >> 16) & 1u);
  unsigned short h = (unsigned short)(r >> 16);
  return __builtin_bit_cast(__bf16, h);
}
__device__ inline float bf2f(__bf16 b) {
  unsigned u = (unsigned)__builtin_bit_cast(unsigned short, b) << 16;
  union { unsigned u; float f; } v; v.u = u;
  return v.f;
}

// ---- async (memory -> LDS) 16-byte copy -----------------------------------
__device__ inline void copy16_g2l(const __bf16* __restrict__ g, __bf16* l) {
#if HAVE_ASYNC_LDS
  __builtin_amdgcn_global_load_async_to_lds_b128(
      (AS1 v4i*)(AS1 void*)g, (AS3 v4i*)(AS3 void*)l, 0, 0);
#else
  *(uint4*)l = *(const uint4*)g;
#endif
}

__device__ inline void wait_async_all() {
#if HAVE_ASYNC_LDS
#  if __has_builtin(__builtin_amdgcn_s_wait_asynccnt)
  __builtin_amdgcn_s_wait_asynccnt(0);
#  else
  asm volatile("s_wait_asynccnt 0x0" ::: "memory");
#  endif
#endif
}

// ---- A fragment: 16x32 bf16, layout per ISA 7.12.2 ------------------------
// lanes 0-15 : M=lane,    v0..v3 K=0..7,   v4..v7 K=16..23
// lanes 16-31: M=lane-16, v0..v3 K=8..15,  v4..v7 K=24..31
__device__ inline v16bf load_a_frag(const __bf16* As_, int lane, int m0, int kk, int stride) {
  const __bf16* row = As_ + (m0 + (lane & 15)) * stride + kk;
  const int kb = (lane >> 4) * 8;
  union { v16bf v; __bf16 e[16]; } u;
#pragma unroll
  for (int i = 0; i < 4; ++i) {
    u.e[2 * i]         = row[kb + 2 * i];
    u.e[2 * i + 1]     = row[kb + 2 * i + 1];
    u.e[8 + 2 * i]     = row[16 + kb + 2 * i];
    u.e[8 + 2 * i + 1] = row[16 + kb + 2 * i + 1];
  }
  return u.v;
}

// ---- B fragment: 32x16 bf16 (K x N); Ws stored W-layout [n][k] ------------
// lanes 0-15 : N=lane,    K=0..7,16..23 (2/VGPR);  lanes 16-31: N=lane-16, K=8..15,24..31
__device__ inline v16bf load_b_frag(const __bf16* Ws_, int lane, int n0, int kk, int stride) {
  const __bf16* row = Ws_ + (n0 + (lane & 15)) * stride + kk;
  const int kb = (lane >> 4) * 8;
  union { v16bf v; __bf16 e[16]; } u;
#pragma unroll
  for (int i = 0; i < 8; ++i) {
    u.e[2 * i]     = row[(i >> 2) * 16 + kb + 2 * (i & 3)];
    u.e[2 * i + 1] = row[(i >> 2) * 16 + kb + 2 * (i & 3) + 1];
  }
  return u.v;
}

// ---------------------------------------------------------------------------
// C[M,N] (+)= A[M,K] * W[N,K]^T,  A/W bf16, C f32 (or bf16 for EPI_RELU_BF16)
// Block: 256 threads = 8 waves in 4(M) x 2(N); wave tile 32x32.
// Double-buffered LDS, async DMA fills, 8 WMMAs per wave per K-step of 64.
// ---------------------------------------------------------------------------
template <int EPI>
__global__ __launch_bounds__(256) void gemm_wmma_bf16(
    const __bf16* __restrict__ A, const __bf16* __restrict__ W,
    void* __restrict__ Cv, int M, int N, int K,
    const float* __restrict__ rowscale, int rs_stride) {
  __shared__ alignas(16) __bf16 As[2][BM * LDAs];
  __shared__ alignas(16) __bf16 Ws[2][BN * LDBs];

  const int tid   = threadIdx.x;
  const int lane  = tid & 31;
  const int wave  = tid >> 5;
  const int waveM = wave & 3;
  const int waveN = wave >> 2;
  const int bm = blockIdx.y * BM;
  const int bn = blockIdx.x * BN;

  v8f acc[2][2] = {};

  auto fill = [&](int k0, int b) {
    // A tile: 128 rows x 64 bf16 = 1024 16B-chunks; 4 per thread
#pragma unroll
    for (int i = 0; i < 4; ++i) {
      int c = i * 256 + tid;
      int r = c >> 3, col = (c & 7) * 8;
      copy16_g2l(A + (size_t)(bm + r) * K + k0 + col, &As[b][r * LDAs + col]);
    }
    // W tile: 64 rows x 64 bf16 = 512 chunks; 2 per thread
#pragma unroll
    for (int i = 0; i < 2; ++i) {
      int c = i * 256 + tid;
      int r = c >> 3, col = (c & 7) * 8;
      copy16_g2l(W + (size_t)(bn + r) * K + k0 + col, &Ws[b][r * LDBs + col]);
    }
  };

  fill(0, 0);
  int buf = 0;
  for (int k0 = 0; k0 < K; k0 += BK) {
    wait_async_all();       // my async fills for buffer `buf` done
    __syncthreads();        // everyone's fills done; everyone done reading buf^1
    if (k0 + BK < K) fill(k0 + BK, buf ^ 1);

    const int am0 = waveM * 32;
    const int wn0 = waveN * 32;
#pragma unroll
    for (int kk = 0; kk < BK; kk += 32) {
      v16bf a0 = load_a_frag(&As[buf][0], lane, am0,      kk, LDAs);
      v16bf a1 = load_a_frag(&As[buf][0], lane, am0 + 16, kk, LDAs);
      v16bf b0 = load_b_frag(&Ws[buf][0], lane, wn0,      kk, LDBs);
      v16bf b1 = load_b_frag(&Ws[buf][0], lane, wn0 + 16, kk, LDBs);
      acc[0][0] = __builtin_amdgcn_wmma_f32_16x16x32_bf16(false, a0, false, b0, (short)0, acc[0][0], false, false);
      acc[0][1] = __builtin_amdgcn_wmma_f32_16x16x32_bf16(false, a0, false, b1, (short)0, acc[0][1], false, false);
      acc[1][0] = __builtin_amdgcn_wmma_f32_16x16x32_bf16(false, a1, false, b0, (short)0, acc[1][0], false, false);
      acc[1][1] = __builtin_amdgcn_wmma_f32_16x16x32_bf16(false, a1, false, b1, (short)0, acc[1][1], false, false);
    }
    buf ^= 1;
  }

  // --- epilogue: C/D layout: m = base + (lane>>4)*8 + i, n = lane&15 -------
  const int mrow0 = bm + waveM * 32;
  const int ncol0 = bn + waveN * 32;
#pragma unroll
  for (int mi = 0; mi < 2; ++mi) {
#pragma unroll
    for (int ni = 0; ni < 2; ++ni) {
      const int n     = ncol0 + ni * 16 + (lane & 15);
      const int mbase = mrow0 + mi * 16 + ((lane >> 4) << 3);
#pragma unroll
      for (int i = 0; i < 8; ++i) {
        const int m = mbase + i;
        const float val = acc[mi][ni][i];
        const size_t idx = (size_t)m * N + n;
        if (EPI == EPI_STORE) {
          ((float*)Cv)[idx] = val;
        } else if (EPI == EPI_RELU_BF16) {
          ((__bf16*)Cv)[idx] = f2bf(fmaxf(val, 0.0f));
        } else if (EPI == EPI_ADD) {
          ((float*)Cv)[idx] += val;
        } else {
          ((float*)Cv)[idx] += rowscale[(size_t)m * rs_stride] * val;
        }
      }
    }
  }
}

// ---------------------------------------------------------------------------
// f32 -> bf16 bulk converter (grid-stride, float4 granularity)
// ---------------------------------------------------------------------------
__global__ __launch_bounds__(256) void cvt_f32_bf16(
    const float* __restrict__ in, __bf16* __restrict__ out, long n) {
  const long stride = (long)gridDim.x * 256 * 4;
  for (long i = ((long)blockIdx.x * 256 + threadIdx.x) * 4; i < n; i += stride) {
    const float4 f = *(const float4*)(in + i);
    out[i]     = f2bf(f.x);
    out[i + 1] = f2bf(f.y);
    out[i + 2] = f2bf(f.z);
    out[i + 3] = f2bf(f.w);
  }
}

// ---------------------------------------------------------------------------
// Embedding: out[row][d] = wte[tok][d] + wpe[t][d]   (f32 out)
// ---------------------------------------------------------------------------
__global__ __launch_bounds__(256) void embed_kernel(
    const int* __restrict__ tokens, const float* __restrict__ wte,
    const float* __restrict__ wpe, float* __restrict__ out) {
  const int row = blockIdx.x;           // b*T + t
  const int t   = row % Tc;
  const int tok = tokens[row];
  for (int d = threadIdx.x; d < Dc; d += 256)
    out[(size_t)row * Dc + d] = wte[(size_t)tok * Dc + d] + wpe[(size_t)t * Dc + d];
}

// ---------------------------------------------------------------------------
// LayerNorm, one block per row; templated f32 or bf16 output
// ---------------------------------------------------------------------------
template <bool BFOUT>
__global__ __launch_bounds__(256) void ln_kernel(
    const float* __restrict__ in, void* __restrict__ outv,
    const float* __restrict__ g, const float* __restrict__ b) {
  __shared__ float red[256];
  const int row = blockIdx.x;
  const int tid = threadIdx.x;
  const float* x = in + (size_t)row * Dc;

  float lsum = 0.f;
  for (int d = tid; d < Dc; d += 256) lsum += x[d];
  red[tid] = lsum; __syncthreads();
  for (int s = 128; s > 0; s >>= 1) { if (tid < s) red[tid] += red[tid + s]; __syncthreads(); }
  const float mu = red[0] / Dc;
  __syncthreads();

  float lvar = 0.f;
  for (int d = tid; d < Dc; d += 256) { float t = x[d] - mu; lvar += t * t; }
  red[tid] = lvar; __syncthreads();
  for (int s = 128; s > 0; s >>= 1) { if (tid < s) red[tid] += red[tid + s]; __syncthreads(); }
  const float rstd = rsqrtf(red[0] / Dc + 1e-5f);

  for (int d = tid; d < Dc; d += 256) {
    const float v = (x[d] - mu) * rstd * g[d] + b[d];
    if (BFOUT) ((__bf16*)outv)[(size_t)row * Dc + d] = f2bf(v);
    else       ((float*)outv)[(size_t)row * Dc + d]  = v;
  }
}

// ---------------------------------------------------------------------------
// Causal attention, one wave per query row; lanes own dims {l, l+32} of 64.
// Online softmax with the reference's -1e30 mask semantics. bf16 output.
// qkv layout: [BT][3D]; q at h*64, k at D+h*64, v at 2D+h*64.
// ---------------------------------------------------------------------------
__global__ __launch_bounds__(128) void attn_kernel(
    const float* __restrict__ qkv, __bf16* __restrict__ out) {
  const int lane = threadIdx.x & 31;
  const int wv   = threadIdx.x >> 5;            // 0..3
  const int t    = blockIdx.x * 4 + wv;
  const int h    = blockIdx.y;
  const int b    = blockIdx.z;

  const float* qrow = qkv + (size_t)(b * Tc + t) * (3 * Dc) + h * HEADc;
  const float q0 = qrow[lane]      * 0.125f;    // 1/sqrt(64)
  const float q1 = qrow[lane + 32] * 0.125f;

  float m = -1e30f, s = 0.f, a0 = 0.f, a1 = 0.f;
  for (int j = 0; j <= t; ++j) {
    const float* krow = qkv + (size_t)(b * Tc + j) * (3 * Dc) + Dc + h * HEADc;
    float part = q0 * krow[lane] + q1 * krow[lane + 32];
#pragma unroll
    for (int off = 16; off > 0; off >>= 1) part += __shfl_xor(part, off, 32);
    const float nm  = fmaxf(m, part);
    const float fac = __expf(m - nm);
    const float p   = __expf(part - nm);
    const float* vrow = qkv + (size_t)(b * Tc + j) * (3 * Dc) + 2 * Dc + h * HEADc;
    s  = s  * fac + p;
    a0 = a0 * fac + p * vrow[lane];
    a1 = a1 * fac + p * vrow[lane + 32];
    m = nm;
  }
  const float inv = 1.0f / s;
  __bf16* orow = out + (size_t)(b * Tc + t) * Dc + h * HEADc;
  orow[lane]      = f2bf(a0 * inv);
  orow[lane + 32] = f2bf(a1 * inv);
}

// ---------------------------------------------------------------------------
// Router: 8 waves = 8 expert dots per token (bf16 h); softmax+top2 -> mp[row][8]
// ---------------------------------------------------------------------------
__global__ __launch_bounds__(256) void router_kernel(
    const __bf16* __restrict__ h, const float* __restrict__ rw,
    float* __restrict__ mp) {
  __shared__ float sc[Ec];
  const int row  = blockIdx.x;
  const int lane = threadIdx.x & 31;
  const int e    = threadIdx.x >> 5;            // 0..7

  const __bf16* x = h  + (size_t)row * Dc;
  const float*  w = rw + (size_t)e   * Dc;
  float part = 0.f;
  for (int d = lane; d < Dc; d += 32) part += bf2f(x[d]) * w[d];
#pragma unroll
  for (int off = 16; off > 0; off >>= 1) part += __shfl_xor(part, off, 32);
  if (lane == 0) sc[e] = part;
  __syncthreads();

  if (threadIdx.x == 0) {
    float smax = sc[0];
    for (int i = 1; i < Ec; ++i) smax = fmaxf(smax, sc[i]);
    float pr[Ec], psum = 0.f;
    for (int i = 0; i < Ec; ++i) { pr[i] = __expf(sc[i] - smax); psum += pr[i]; }
    int i0 = 0;
    for (int i = 1; i < Ec; ++i) if (sc[i] > sc[i0]) i0 = i;
    int i1 = -1;
    for (int i = 0; i < Ec; ++i) if (i != i0 && (i1 < 0 || sc[i] > sc[i1])) i1 = i;
    const float w0 = pr[i0] / psum, w1 = pr[i1] / psum;
    const float tot = w0 + w1 + 1e-8f;
    float* o = mp + (size_t)row * Ec;
    for (int i = 0; i < Ec; ++i) o[i] = 0.f;
    o[i0] = w0 / tot;
    o[i1] = w1 / tot;
  }
}

// ---------------------------------------------------------------------------
extern "C" void kernel_launch(void* const* d_in, const int* in_sizes, int n_in,
                              void* d_out, int out_size, void* d_ws, size_t ws_size,
                              hipStream_t stream) {
  const int*   tokens   = (const int*)  d_in[0];
  const float* wte      = (const float*)d_in[1];
  const float* wpe      = (const float*)d_in[2];
  const float* n0g      = (const float*)d_in[3];
  const float* n0b      = (const float*)d_in[4];
  const float* n1g      = (const float*)d_in[5];
  const float* n1b      = (const float*)d_in[6];
  const float* n2g      = (const float*)d_in[7];
  const float* n2b      = (const float*)d_in[8];
  const float* qkv_w    = (const float*)d_in[9];
  const float* proj_w   = (const float*)d_in[10];
  const float* router_w = (const float*)d_in[11];
  const float* fc1      = (const float*)d_in[12];
  const float* fc2      = (const float*)d_in[13];
  const float* lm_w     = (const float*)d_in[14];
  float* out = (float*)d_out;

  // ---- workspace carve-up (256B-aligned regions) ----
  char* p = (char*)d_ws;
  auto carve = [&](size_t bytes) { void* r = (void*)p; p += (bytes + 255) & ~(size_t)255; return r; };
  float*  x     = (float*) carve((size_t)BT * Dc * 4);        // residual stream
  float*  qkvb  = (float*) carve((size_t)BT * 3 * Dc * 4);    // qkv (also embed temp)
  float*  mp    = (float*) carve((size_t)BT * Ec * 4);        // routing weights
  __bf16* hbf   = (__bf16*)carve((size_t)BT * Dc * 2);        // LN output (GEMM A)
  __bf16* attbf = (__bf16*)carve((size_t)BT * Dc * 2);        // attention out
  __bf16* hidbf = (__bf16*)carve((size_t)BT * DFFc * 2);      // expert hidden
  __bf16* xbf   = (__bf16*)carve((size_t)BT * Dc * 2);        // x for lm head
  __bf16* wqkv  = (__bf16*)carve((size_t)Lc * 3 * Dc * Dc * 2);
  __bf16* wproj = (__bf16*)carve((size_t)Lc * Dc * Dc * 2);
  __bf16* wfc1  = (__bf16*)carve((size_t)Lc * Ec * DFFc * Dc * 2);
  __bf16* wfc2  = (__bf16*)carve((size_t)Lc * Ec * Dc * DFFc * 2);
  __bf16* wlm   = (__bf16*)carve((size_t)Vc * Dc * 2);

  const dim3 blk256(256), blk128(128);

  // ---- one-shot weight conversion to bf16 ----
  cvt_f32_bf16<<<2048, blk256, 0, stream>>>(qkv_w,  wqkv,  (long)Lc * 3 * Dc * Dc);
  cvt_f32_bf16<<<2048, blk256, 0, stream>>>(proj_w, wproj, (long)Lc * Dc * Dc);
  cvt_f32_bf16<<<4096, blk256, 0, stream>>>(fc1,    wfc1,  (long)Lc * Ec * DFFc * Dc);
  cvt_f32_bf16<<<4096, blk256, 0, stream>>>(fc2,    wfc2,  (long)Lc * Ec * Dc * DFFc);
  cvt_f32_bf16<<<2048, blk256, 0, stream>>>(lm_w,   wlm,   (long)Vc * Dc);

  // ---- x = LN0(wte[tok] + wpe) ----
  embed_kernel<<<BT, blk256, 0, stream>>>(tokens, wte, wpe, qkvb);
  ln_kernel<false><<<BT, blk256, 0, stream>>>(qkvb, x, n0g, n0b);

  for (int l = 0; l < Lc; ++l) {
    // ---- attention block ----
    ln_kernel<true><<<BT, blk256, 0, stream>>>(x, hbf, n1g + l * Dc, n1b + l * Dc);
    gemm_wmma_bf16<EPI_STORE><<<dim3(3 * Dc / BN, BT / BM), blk256, 0, stream>>>(
        hbf, wqkv + (size_t)l * 3 * Dc * Dc, qkvb, BT, 3 * Dc, Dc, nullptr, 0);
    attn_kernel<<<dim3(Tc / 4, Hc, Bc), blk128, 0, stream>>>(qkvb, attbf);
    gemm_wmma_bf16<EPI_ADD><<<dim3(Dc / BN, BT / BM), blk256, 0, stream>>>(
        attbf, wproj + (size_t)l * Dc * Dc, x, BT, Dc, Dc, nullptr, 0);

    // ---- MoE block (dense experts, routed mixing weights) ----
    ln_kernel<true><<<BT, blk256, 0, stream>>>(x, hbf, n2g + l * Dc, n2b + l * Dc);
    router_kernel<<<BT, blk256, 0, stream>>>(hbf, router_w + (size_t)l * Ec * Dc, mp);
    for (int e = 0; e < Ec; ++e) {
      gemm_wmma_bf16<EPI_RELU_BF16><<<dim3(DFFc / BN, BT / BM), blk256, 0, stream>>>(
          hbf, wfc1 + ((size_t)l * Ec + e) * DFFc * Dc, hidbf, BT, DFFc, Dc, nullptr, 0);
      gemm_wmma_bf16<EPI_SCALE_ADD><<<dim3(Dc / BN, BT / BM), blk256, 0, stream>>>(
          hidbf, wfc2 + ((size_t)l * Ec + e) * Dc * DFFc, x, BT, Dc, DFFc, mp + e, Ec);
    }
  }

  // ---- logits = x @ lm_head^T ----
  cvt_f32_bf16<<<1024, blk256, 0, stream>>>(x, xbf, (long)BT * Dc);
  gemm_wmma_bf16<EPI_STORE><<<dim3(Vc / BN, BT / BM), blk256, 0, stream>>>(
      xbf, wlm, out, BT, Vc, Dc, nullptr, 0);
}
